// SmoothAttention_58961311039940
// MI455X (gfx1250) — matrix operand order repaired and checked by reading, not compiled, not run on hardware
//
#include <hip/hip_runtime.h>
#include <hip/hip_bf16.h>
#include <stdint.h>

// ---------------------------------------------------------------------------
// Problem constants (from reference): x[B,T,C] attention, 16 heads, d=64
// ---------------------------------------------------------------------------
#define DIM   1024
#define HEADS 16
#define HD    64          // head dim
#define BB    4           // batch
#define TT    2048        // sequence
#define MM    (BB * TT)   // 8192 rows for the projections

typedef __attribute__((ext_vector_type(16))) __bf16        v16bf;
typedef __attribute__((ext_vector_type(8)))  float         v8f;
typedef __attribute__((ext_vector_type(4)))  unsigned int  v4u;   // 128-bit chunk
typedef __attribute__((ext_vector_type(8)))  unsigned int  v8u;   // 256-bit (8 SGPRs)

// bf16 round-to-nearest-even from fp32
__device__ __forceinline__ unsigned short f2bf(float f) {
  union { float f; unsigned int u; } x; x.f = f;
  unsigned int u = x.u;
  unsigned int r = u + 0x7fffu + ((u >> 16) & 1u);
  return (unsigned short)(r >> 16);
}

// One WMMA 16x32 bf16 fragment = 16 bf16 per lane = two 128-bit chunks
struct Frag {
  union { v4u q[2]; v16bf v; };
};

// ---------------------------------------------------------------------------
// fp32 -> bf16 cast (grid-stride)
// ---------------------------------------------------------------------------
__global__ void cast_bf16_kernel(const float* __restrict__ in,
                                 unsigned short* __restrict__ out, int n) {
  int i = blockIdx.x * blockDim.x + threadIdx.x;
  int stride = gridDim.x * blockDim.x;
  for (; i < n; i += stride) out[i] = f2bf(in[i]);
}

// ---------------------------------------------------------------------------
// Y = X * W^T + bias.   X: [M,K] bf16 row-major.  W: [N,K] bf16 row-major.
// One wave -> one 16x64 tile of Y (4 accumulators), K-loop step 32.
// The A fragment is reused across 4 WMMAs per K step (2 A loads + 8 B loads
// per 4 matrix ops). X+W in bf16 fit in the 192MB L2, so B re-reads are L2 hits.
// ---------------------------------------------------------------------------
template<bool BF16OUT>
__global__ void __launch_bounds__(256)
gemm_bias_wmma(const unsigned short* __restrict__ A,
               const unsigned short* __restrict__ W,
               const float* __restrict__ bias,
               unsigned short* __restrict__ outB,
               float* __restrict__ outF,
               int M, int N, int K) {
  const int lane = threadIdx.x & 31;
  const int wave = threadIdx.x >> 5;
  const int lo = lane & 15, hi = lane >> 4;
  const int tiles_n = N >> 6;                       // 64-wide N tiles
  const int id = blockIdx.x * (blockDim.x >> 5) + wave;
  const int mt = id / tiles_n, nt = id % tiles_n;
  if (mt >= (M >> 4)) return;                       // wave-uniform guard
  const int m0 = mt << 4, n0 = nt << 6;

  const unsigned short* arow = A + (size_t)(m0 + lo) * K + hi * 8;
  const unsigned short* brow0 = W + (size_t)(n0 + lo) * K + hi * 8;

  v8f c[4];
  #pragma unroll
  for (int j = 0; j < 4; ++j)
    #pragma unroll
    for (int r = 0; r < 8; ++r) c[j][r] = 0.0f;

  for (int kk = 0; kk < K; kk += 32) {
    Frag a;
    a.q[0] = *(const v4u*)(arow + kk);
    a.q[1] = *(const v4u*)(arow + kk + 16);
    __builtin_prefetch(arow + kk + 64, 0, 0);       // global_prefetch_b8
    #pragma unroll
    for (int j = 0; j < 4; ++j) {
      Frag b;
      const unsigned short* brow = brow0 + (size_t)(j * 16) * K + kk;
      b.q[0] = *(const v4u*)(brow);
      b.q[1] = *(const v4u*)(brow + 16);
      c[j] = __builtin_amdgcn_wmma_f32_16x16x32_bf16(false, a.v, false, b.v,
                                                     (short)0, c[j],
                                                     false, false);
    }
  }

  #pragma unroll
  for (int j = 0; j < 4; ++j) {
    const int col = n0 + j * 16 + lo;
    const float bs = bias[col];
    #pragma unroll
    for (int r = 0; r < 8; ++r) {
      const int row = m0 + r + 8 * hi;
      const float val = c[j][r] + bs;
      if (BF16OUT) outB[(size_t)row * N + col] = f2bf(val);
      else         outF[(size_t)row * N + col] = val;
    }
  }
}

// ---------------------------------------------------------------------------
// Tensor Data Mover: build D# groups in SGPR tuples and issue
// tensor_load_to_lds for a 2-D tile: tile_dim0=64 (bf16 cols), tile_dim1=32
// rows, row stride = DIM elements, data_size=2B. Group0 carries lds/global
// address + type=2; group1 carries dims/strides. 2-group form (VADDR2/3 NULL).
// ---------------------------------------------------------------------------
__device__ __forceinline__ void tdm_load_tile_32x64(const unsigned short* gsrc,
                                                    unsigned short* lds_dst) {
  const unsigned long long ga = (unsigned long long)(uintptr_t)gsrc;
  const unsigned int lds_off = (unsigned int)(uintptr_t)lds_dst; // LDS aperture low bits
  const v4u g0 = { 1u,                                   // count=1, user mode
                   lds_off,                              // lds_addr
                   (unsigned int)ga,                     // global_addr[31:0]
                   (unsigned int)((ga >> 32) & 0x01FFFFFFu) | 0x80000000u }; // addr[56:32] | type=2
  const v8u g1 = { 0x00010000u,                          // data_size=1 (2 bytes)
                   (4096u << 16),                        // tensor_dim0[15:0] in [31:16]
                   (8192u << 16),                        // tensor_dim0[31:16]=0, tensor_dim1[15:0]
                   (64u << 16),                          // tensor_dim1[31:16]=0, tile_dim0=64
                   32u,                                  // tile_dim1=32, tile_dim2=0
                   (unsigned int)DIM,                    // tensor_dim0_stride[31:0]
                   0u,                                   // stride[47:32], dim1_stride[15:0]
                   0u };
  asm volatile("tensor_load_to_lds %0, %1" :: "s"(g0), "s"(g1) : "memory");
}

// ---------------------------------------------------------------------------
// Flash attention: one block = 4 waves = 64 query rows of one (b,h).
// K/V tiles staged into LDS by the TDM (wave 0 issues, TENSORcnt drains,
// block barrier releases); V transposed LDS->LDS; online softmax in registers.
// ---------------------------------------------------------------------------
__global__ void __launch_bounds__(128)
flash_attn_kernel(const unsigned short* __restrict__ Qm,
                  const unsigned short* __restrict__ Km,
                  const unsigned short* __restrict__ Vm,
                  unsigned short* __restrict__ Om) {
  // [0,2048)=Kt [key][d] | [2048,4096)=Vrow [key][d] | [4096,6144)=Vt [d][key]
  // | [6144,8192)=Pb per-wave P scratch
  __shared__ alignas(16) unsigned short smem[8192];
  unsigned short* Kt   = smem;
  unsigned short* Vrow = smem + 2048;
  unsigned short* Vt   = smem + 4096;
  unsigned short* Pb   = smem + 6144;

  const int tid  = threadIdx.x;
  const int lane = tid & 31, w = tid >> 5;
  const int lo = lane & 15, hi = lane >> 4;

  const int qblocks = TT / 64;
  const int qb = blockIdx.x % qblocks;
  const int bh = blockIdx.x / qblocks;
  const int h  = bh % HEADS, bat = bh / HEADS;
  const int q0 = qb * 64 + w * 16;
  const int c0 = h * HD;
  const float scale = 0.125f;                 // HD^-0.5

  // Q 16x64 -> two 16x32 A fragments, loaded once
  Frag aq[2];
  {
    const unsigned short* qrow =
        Qm + (size_t)(bat * TT + q0 + lo) * DIM + c0 + hi * 8;
    aq[0].q[0] = *(const v4u*)(qrow);
    aq[0].q[1] = *(const v4u*)(qrow + 16);
    aq[1].q[0] = *(const v4u*)(qrow + 32);
    aq[1].q[1] = *(const v4u*)(qrow + 48);
  }

  float m_r[8], l_r[8];
  v8f o[4];
  #pragma unroll
  for (int r = 0; r < 8; ++r) { m_r[r] = -3.0e38f; l_r[r] = 0.0f; }
  #pragma unroll
  for (int f = 0; f < 4; ++f)
    #pragma unroll
    for (int r = 0; r < 8; ++r) o[f][r] = 0.0f;

  for (int t0 = 0; t0 < TT; t0 += 32) {
    // ---- TDM stage: K and V 32x64 tiles -> LDS (wave 0 issues) ----
    if (w == 0) {
      tdm_load_tile_32x64(Km + (size_t)(bat * TT + t0) * DIM + c0, Kt);
      tdm_load_tile_32x64(Vm + (size_t)(bat * TT + t0) * DIM + c0, Vrow);
      __builtin_amdgcn_s_wait_tensorcnt(0);
    }
    __syncthreads();

    // ---- transpose V in LDS: Vrow[key][d] -> Vt[d][key] ----
    {
      const int r  = tid >> 2;
      const int cb = (tid & 3) * 16;
      union { v4u q[2]; unsigned short s[16]; } tmp;
      tmp.q[0] = *(const v4u*)&Vrow[r * 64 + cb];
      tmp.q[1] = *(const v4u*)&Vrow[r * 64 + cb + 8];
      #pragma unroll
      for (int i = 0; i < 16; ++i) Vt[(cb + i) * 32 + r] = tmp.s[i];
    }
    __syncthreads();

    // ---- S = Q * K^T (16x32), 4 WMMAs ----
    v8f s[2];
    #pragma unroll
    for (int j = 0; j < 2; ++j) {
      #pragma unroll
      for (int r = 0; r < 8; ++r) s[j][r] = 0.0f;
      #pragma unroll
      for (int kidx = 0; kidx < 2; ++kidx) {
        Frag bk;
        const unsigned short* kp = &Kt[(j * 16 + lo) * 64 + kidx * 32 + hi * 8];
        bk.q[0] = *(const v4u*)(kp);
        bk.q[1] = *(const v4u*)(kp + 16);
        s[j] = __builtin_amdgcn_wmma_f32_16x16x32_bf16(false, aq[kidx].v, false,
                                                       bk.v, (short)0, s[j],
                                                       false, false);
      }
      #pragma unroll
      for (int r = 0; r < 8; ++r) s[j][r] *= scale;
    }

    // ---- online softmax over 32 keys (rows live across 16-lane halves) ----
    float p0[8], p1[8], alpha[8];
    #pragma unroll
    for (int r = 0; r < 8; ++r) {
      float t = fmaxf(s[0][r], s[1][r]);
      #pragma unroll
      for (int off = 1; off < 16; off <<= 1) t = fmaxf(t, __shfl_xor(t, off));
      const float mnew = fmaxf(m_r[r], t);
      alpha[r] = __expf(m_r[r] - mnew);
      p0[r] = __expf(s[0][r] - mnew);
      p1[r] = __expf(s[1][r] - mnew);
      float rs = p0[r] + p1[r];
      #pragma unroll
      for (int off = 1; off < 16; off <<= 1) rs += __shfl_xor(rs, off);
      l_r[r] = l_r[r] * alpha[r] + rs;
      m_r[r] = mnew;
    }
    #pragma unroll
    for (int f = 0; f < 4; ++f)
      #pragma unroll
      for (int r = 0; r < 8; ++r) o[f][r] *= alpha[r];

    // ---- P: C/D-layout -> A-layout via per-wave LDS round-trip ----
    {
      unsigned short* pw = &Pb[w * 512];
      #pragma unroll
      for (int r = 0; r < 8; ++r) {
        const int row = r + 8 * hi;
        pw[row * 32 + lo]      = f2bf(p0[r]);
        pw[row * 32 + 16 + lo] = f2bf(p1[r]);
      }
    }
    asm volatile("s_wait_dscnt 0" ::: "memory");   // per-wave RAW on LDS

    Frag ap;
    {
      const unsigned short* pr = &Pb[w * 512 + lo * 32 + hi * 8];
      ap.q[0] = *(const v4u*)(pr);
      ap.q[1] = *(const v4u*)(pr + 16);
    }

    // ---- O += P * V (16x64), 4 WMMAs against transposed V tile ----
    #pragma unroll
    for (int dt = 0; dt < 4; ++dt) {
      Frag bv;
      const unsigned short* vp = &Vt[(dt * 16 + lo) * 32 + hi * 8];
      bv.q[0] = *(const v4u*)(vp);
      bv.q[1] = *(const v4u*)(vp + 16);
      o[dt] = __builtin_amdgcn_wmma_f32_16x16x32_bf16(false, ap.v, false, bv.v,
                                                      (short)0, o[dt],
                                                      false, false);
    }
    __syncthreads();
  }

  // ---- normalize and write O as bf16 [B,T,DIM] ----
  float inv[8];
  #pragma unroll
  for (int r = 0; r < 8; ++r) inv[r] = 1.0f / l_r[r];
  #pragma unroll
  for (int f = 0; f < 4; ++f) {
    #pragma unroll
    for (int r = 0; r < 8; ++r) {
      const int row = q0 + r + 8 * hi;
      Om[(size_t)(bat * TT + row) * DIM + c0 + f * 16 + lo] =
          f2bf(o[f][r] * inv[r]);
    }
  }
}

// ---------------------------------------------------------------------------
// Host-side orchestration (all on `stream`; scratch only in d_ws)
// ---------------------------------------------------------------------------
extern "C" void kernel_launch(void* const* d_in, const int* in_sizes, int n_in,
                              void* d_out, int out_size, void* d_ws, size_t ws_size,
                              hipStream_t stream) {
  const float* x  = (const float*)d_in[0];
  const float* Wq = (const float*)d_in[1];
  const float* bq = (const float*)d_in[2];
  const float* Wk = (const float*)d_in[3];
  const float* bk = (const float*)d_in[4];
  const float* Wv = (const float*)d_in[5];
  const float* bv = (const float*)d_in[6];
  const float* Wp = (const float*)d_in[7];
  const float* bp = (const float*)d_in[8];

  unsigned short* ws = (unsigned short*)d_ws;
  const size_t MK = (size_t)MM * DIM;   // 8388608
  const size_t WK = (size_t)DIM * DIM;  // 1048576
  unsigned short* Xb  = ws;
  unsigned short* Wqb = Xb  + MK;
  unsigned short* Wkb = Wqb + WK;
  unsigned short* Wvb = Wkb + WK;
  unsigned short* Wpb = Wvb + WK;
  unsigned short* Qb  = Wpb + WK;
  unsigned short* Kb  = Qb  + MK;
  unsigned short* Vb  = Kb  + MK;
  unsigned short* Ob  = Vb  + MK;

  // 1) casts to bf16
  cast_bf16_kernel<<<2048, 256, 0, stream>>>(x,  Xb,  (int)MK);
  cast_bf16_kernel<<<1024, 256, 0, stream>>>(Wq, Wqb, (int)WK);
  cast_bf16_kernel<<<1024, 256, 0, stream>>>(Wk, Wkb, (int)WK);
  cast_bf16_kernel<<<1024, 256, 0, stream>>>(Wv, Wvb, (int)WK);
  cast_bf16_kernel<<<1024, 256, 0, stream>>>(Wp, Wpb, (int)WK);

  // 2) Q/K/V projections (bf16 out): 16x64 tile per wave
  const int tiles  = (MM / 16) * (DIM / 64);   // 8192 wave tiles
  const int blocks = tiles / 8;                // 8 waves / 256-thread block
  gemm_bias_wmma<true><<<blocks, 256, 0, stream>>>(Xb, Wqb, bq, Qb, nullptr, MM, DIM, DIM);
  gemm_bias_wmma<true><<<blocks, 256, 0, stream>>>(Xb, Wkb, bk, Kb, nullptr, MM, DIM, DIM);
  gemm_bias_wmma<true><<<blocks, 256, 0, stream>>>(Xb, Wvb, bv, Vb, nullptr, MM, DIM, DIM);

  // 3) flash attention: B*H*(T/64) blocks, 4 waves each
  flash_attn_kernel<<<BB * HEADS * (TT / 64), 128, 0, stream>>>(Qb, Kb, Vb, Ob);

  // 4) output projection (fp32 out + bias)
  gemm_bias_wmma<false><<<blocks, 256, 0, stream>>>(Ob, Wpb, bp, nullptr,
                                                    (float*)d_out, MM, DIM, DIM);
}
